// TestBinary2_90434831385368
// MI455X (gfx1250) — compile-verified, matrix operand order checked by reading, not run on hardware
//
#include <hip/hip_runtime.h>
#include <hip/hip_bf16.h>

// out = x @ sign(W):  M=8192, N=2048, K=2048
// bf16 WMMA path: sign(W) is exactly representable in bf16 (+-1.0),
// x -> bf16 RNE, accumulate in f32 via v_wmma_f32_16x16x32_bf16.

typedef __attribute__((ext_vector_type(16))) __bf16 v16bf;
typedef __attribute__((ext_vector_type(8)))  float  v8f;

#define TOKENS 8192
#define D_IN   2048
#define UNITS  2048

#define BM 128
#define BN 128
#define BK 32

__device__ __forceinline__ unsigned short f32_to_bf16_rne(float f) {
    unsigned int u = __float_as_uint(f);
    unsigned int r = u + 0x7FFFu + ((u >> 16) & 1u);   // round to nearest even
    return (unsigned short)(r >> 16);
}

union FragU { v16bf v; uint4 q[2]; };

__global__ __launch_bounds__(256)
void bgemm_sign_bf16(const float* __restrict__ X,
                     const float* __restrict__ W,
                     float* __restrict__ Out) {
    // A tile [m][k] as bf16, B tile stored transposed [n][k] as bf16 (+-1)
    __shared__ unsigned short As[BM * BK];   // 8 KB
    __shared__ unsigned short Bs[BN * BK];   // 8 KB

    const int tid   = threadIdx.x;
    const int lane  = tid & 31;
    const int wave  = tid >> 5;        // 0..7
    const int waveM = wave & 3;        // 4 waves along M (32 rows each)
    const int waveN = wave >> 2;       // 2 waves along N (64 cols each)
    const int half  = lane >> 4;       // 0/1 (lane group)
    const int l16   = lane & 15;

    const int blockM = blockIdx.y * BM;
    const int blockN = blockIdx.x * BN;

    // x-tile loader: 2 threads per row, 16 floats each
    const int arow = tid >> 1;             // 0..127
    const int acol = (tid & 1) * 16;       // 0 or 16
    // W-tile loader: 8 threads per k-row, 16 floats each (coalesced along n)
    const int wk = tid >> 3;               // 0..31
    const int wn = (tid & 7) * 16;         // 0..112

    v8f acc[2][4];
    {
        v8f z = {};
        #pragma unroll
        for (int i = 0; i < 2; ++i)
            #pragma unroll
            for (int j = 0; j < 4; ++j)
                acc[i][j] = z;
    }

    for (int kb = 0; kb < D_IN; kb += BK) {
        // ---- global loads (b128) issued before the barrier ----
        float4 av[4];
        const float4* xp = (const float4*)(X + (size_t)(blockM + arow) * D_IN + kb + acol);
        #pragma unroll
        for (int j = 0; j < 4; ++j) av[j] = xp[j];

        float4 wv[4];
        const float4* wp = (const float4*)(W + (size_t)(kb + wk) * UNITS + blockN + wn);
        #pragma unroll
        for (int j = 0; j < 4; ++j) wv[j] = wp[j];

        __syncthreads();   // previous iteration's LDS reads complete

        // ---- stage x tile: f32 -> bf16, row-major [m][k] ----
        {
            unsigned short tmp[16];
            #pragma unroll
            for (int j = 0; j < 4; ++j) {
                tmp[j * 4 + 0] = f32_to_bf16_rne(av[j].x);
                tmp[j * 4 + 1] = f32_to_bf16_rne(av[j].y);
                tmp[j * 4 + 2] = f32_to_bf16_rne(av[j].z);
                tmp[j * 4 + 3] = f32_to_bf16_rne(av[j].w);
            }
            uint4* dst = (uint4*)&As[arow * BK + acol];
            dst[0] = *(const uint4*)&tmp[0];
            dst[1] = *(const uint4*)&tmp[8];
        }
        // ---- stage W tile: sign -> +-1 bf16, transposed [n][k] ----
        {
            #pragma unroll
            for (int j = 0; j < 4; ++j) {
                const float f[4] = { wv[j].x, wv[j].y, wv[j].z, wv[j].w };
                #pragma unroll
                for (int e = 0; e < 4; ++e) {
                    const int n = wn + j * 4 + e;
                    Bs[n * BK + wk] = (f[e] >= 0.0f) ? (unsigned short)0x3F80u   // +1.0 bf16
                                                     : (unsigned short)0xBF80u;  // -1.0 bf16
                }
            }
        }
        __syncthreads();

        // ---- fragment loads: two aligned ds_load_b128 per matrix tile ----
        // 16-bit A layout (ISA 7.12.2): lane holds row M=l16 (half selects K
        // groups): K in {8h..8h+7} (VGPR0-3) and {16+8h..16+8h+7} (VGPR4-7).
        FragU a[2], b[4];
        #pragma unroll
        for (int i = 0; i < 2; ++i) {
            const int m = waveM * 32 + i * 16 + l16;
            const uint4* src = (const uint4*)&As[m * BK + half * 8];
            a[i].q[0] = src[0];   // K = 8h .. 8h+7
            a[i].q[1] = src[2];   // K = 16+8h .. 16+8h+7
        }
        #pragma unroll
        for (int j = 0; j < 4; ++j) {
            const int n = waveN * 64 + j * 16 + l16;
            const uint4* src = (const uint4*)&Bs[n * BK + half * 8];
            b[j].q[0] = src[0];
            b[j].q[1] = src[2];
        }

        // ---- 8 independent WMMAs per K-step ----
        #pragma unroll
        for (int i = 0; i < 2; ++i)
            #pragma unroll
            for (int j = 0; j < 4; ++j)
                acc[i][j] = __builtin_amdgcn_wmma_f32_16x16x32_bf16(
                    false, a[i].v, false, b[j].v,
                    (short)0, acc[i][j], false, false);
    }

    // ---- store: C/D layout lane L, VGPR r -> row 8*(L/16)+r, col L%16 ----
    #pragma unroll
    for (int i = 0; i < 2; ++i) {
        #pragma unroll
        for (int j = 0; j < 4; ++j) {
            const int col = blockN + waveN * 64 + j * 16 + l16;
            #pragma unroll
            for (int r = 0; r < 8; ++r) {
                const int row = blockM + waveM * 32 + i * 16 + half * 8 + r;
                Out[(size_t)row * UNITS + col] = acc[i][j][r];
            }
        }
    }
}

extern "C" void kernel_launch(void* const* d_in, const int* in_sizes, int n_in,
                              void* d_out, int out_size, void* d_ws, size_t ws_size,
                              hipStream_t stream) {
    const float* X = (const float*)d_in[0];
    const float* W = (const float*)d_in[1];
    float* Out = (float*)d_out;

    dim3 grid(UNITS / BN, TOKENS / BM);   // (16, 64) -> 1024 workgroups
    dim3 block(256);                      // 8 wave32s per WGP workgroup
    hipLaunchKernelGGL(bgemm_sign_bf16, grid, block, 0, stream, X, W, Out);
}
